// HyperbolicKuramotoAttentionV2_137438954438
// MI455X (gfx1250) — matrix operand order repaired and checked by reading, not compiled, not run on hardware
//
#include <hip/hip_runtime.h>
#include <hip/hip_bf16.h>
#include <math.h>

// HyperbolicKuramotoAttentionV2 for gfx1250 (MI455X), wave32, WMMA f16->f32,
// async global->LDS staging (double-buffered) for shared K/V tiles.
//
// Shapes: B=4, L=1024, D=1024, H=16, Dh=64.
// Outputs (concat in d_out, f32): output[4,1024,1024], phases[4,16], order[4].
//
// Workspace layout (bytes):
//   Qf16  [B,H,L,Dh]  f16 : 0        .. 8 MiB
//   Kf16  [B,H,L,Dh]  f16 : 8 MiB    .. 16 MiB
//   Vtf16 [B,H,Dh,L]  f16 : 16 MiB   .. 24 MiB
//   Ctx   [B,L,D]     f16 : 24 MiB   .. 32 MiB
//   qn    [B,H,L]     f32 : 32 MiB   .. +256 KiB
//   kn    [B,H,L]     f32 : +256 KiB .. +512 KiB
//   coher [B,H]       f32 : +512 KiB .. +512 KiB + 256 B

typedef __attribute__((ext_vector_type(16))) _Float16 v16h;
typedef __attribute__((ext_vector_type(8)))  float    v8f;
typedef __attribute__((ext_vector_type(4)))  int      v4i;

#define TWO_PI_F 6.28318530717958647692f

// Padded LDS row strides (in halves) to avoid bank conflicts on B-layout reads.
#define KS 72   // K tile rows: 32 x (64 data + 8 pad)
#define VS 40   // V tile rows: 64 x (32 data + 8 pad)
#define PS 36   // P tile rows: 16 x (32 data + 4 pad), per wave

#if defined(__HIP_DEVICE_COMPILE__) && __has_builtin(__builtin_amdgcn_global_load_async_to_lds_b128)
#define HK_ASYNC_LDS 1
#else
#define HK_ASYNC_LDS 0
#endif

// Copy one 16-byte chunk global -> LDS. Async path uses the CDNA5
// GLOBAL_LOAD_ASYNC_TO_LDS_B128 instruction (ASYNCcnt-tracked).
// Builtin signature (from hipcc diagnostic): arg0 = v4i AS(1)* (non-const),
// arg1 = LDS destination, arg2/arg3 = imm offset / cpol.
__device__ __forceinline__ void hk_cp16(const _Float16* __restrict__ g,
                                        _Float16* __restrict__ l) {
#if HK_ASYNC_LDS
    __builtin_amdgcn_global_load_async_to_lds_b128(
        (__attribute__((address_space(1))) v4i*)g,
        (__attribute__((address_space(3))) v4i*)l,
        0, 0);
#else
    *(float4*)l = *(const float4*)g;
#endif
}

__device__ __forceinline__ void hk_async_wait() {
#if HK_ASYNC_LDS
#if __has_builtin(__builtin_amdgcn_s_wait_asynccnt)
    __builtin_amdgcn_s_wait_asynccnt(0);
#else
    asm volatile("s_wait_asynccnt 0x0" ::: "memory");
#endif
#endif
}

// ---------------------------------------------------------------------------
// Kuramoto phase step (tiny): one block, 64 threads = (b,h) pairs.
// ---------------------------------------------------------------------------
__global__ void hk_phase_kernel(const float* __restrict__ base,
                                const float* __restrict__ natf,
                                const float* __restrict__ ph0,
                                float* __restrict__ out_phases,   // [4,16]
                                float* __restrict__ out_order,    // [4]
                                float* __restrict__ coher) {      // [4,16]
    int t = threadIdx.x;
    if (t >= 64) return;
    int b = t >> 4, h = t & 15;

    float newph[16];
    for (int j = 0; j < 16; ++j) {
        float mx = -1e30f;
        for (int q = 0; q < 16; ++q) mx = fmaxf(mx, base[j * 16 + q]);
        float e[16]; float se = 0.f;
        for (int q = 0; q < 16; ++q) { e[q] = expf(base[j * 16 + q] - mx); se += e[q]; }
        float th_j = ph0[j];
        float csum = 0.f;
        for (int q = 0; q < 16; ++q) csum += (e[q] / se) * sinf(th_j - ph0[q]);
        float dph = natf[j] + (1.0f / 16.0f) * csum;   // COUPLING_STRENGTH/H
        newph[j] = fmodf(th_j + 0.1f * dph, TWO_PI_F); // DT=0.1
    }

    float coh = 0.f;
    for (int j = 0; j < 16; ++j) coh += cosf(newph[h] - newph[j]);
    coh *= (1.0f / 16.0f);
    coher[t] = coh;
    out_phases[t] = newph[h];

    if (h == 0) {
        float c = 0.f, s = 0.f;
        for (int j = 0; j < 16; ++j) { c += cosf(newph[j]); s += sinf(newph[j]); }
        c *= (1.0f / 16.0f); s *= (1.0f / 16.0f);
        out_order[b] = sqrtf(c * c + s * s);
    }
}

// ---------------------------------------------------------------------------
// QKV projection GEMM: C[m,n] = sum_k X[m,k] * W[n,k] + bias[n]
// block = 128 threads (4 waves); wave computes a 16(M)x64(N) C tile.
// blockIdx.z selects Q / K / V; V is stored transposed [b,h,d,l].
// ---------------------------------------------------------------------------
__global__ void hk_gemm_qkv(const float* __restrict__ X,
                            const float* __restrict__ Wq, const float* __restrict__ bq,
                            const float* __restrict__ Wk, const float* __restrict__ bk,
                            const float* __restrict__ Wv, const float* __restrict__ bv,
                            _Float16* __restrict__ Qo,
                            _Float16* __restrict__ Ko,
                            _Float16* __restrict__ Vto) {
    const int mode = blockIdx.z;
    const float* W; const float* bias; _Float16* out;
    if (mode == 0)      { W = Wq; bias = bq; out = Qo;  }
    else if (mode == 1) { W = Wk; bias = bk; out = Ko;  }
    else                { W = Wv; bias = bv; out = Vto; }

    const int wave = threadIdx.x >> 5;
    const int lane = threadIdx.x & 31;
    const int lm = lane & 15, lh = lane >> 4;
    const int m0 = blockIdx.y * 64 + wave * 16;
    const int n0 = blockIdx.x * 64;

    v8f c[4];
    for (int t = 0; t < 4; ++t)
        for (int j = 0; j < 8; ++j) c[t][j] = 0.0f;

    for (int k0 = 0; k0 < 1024; k0 += 32) {
        v16h a;
        const float* px = X + (size_t)(m0 + lm) * 1024 + k0 + lh * 16;
        for (int i = 0; i < 8; ++i) {
            a[i]     = (_Float16)px[i];
            a[8 + i] = (_Float16)px[8 + i];
        }
        for (int t = 0; t < 4; ++t) {
            v16h bm;
            const float* pw = W + (size_t)(n0 + t * 16 + lm) * 1024 + k0 + lh * 16;
            for (int i = 0; i < 16; ++i) bm[i] = (_Float16)pw[i];
            c[t] = __builtin_amdgcn_wmma_f32_16x16x32_f16(
                false, a, false, bm, (short)0, c[t], false, false);
        }
    }

    for (int t = 0; t < 4; ++t) {
        const int n = n0 + t * 16 + lm;
        const int hh = n >> 6, d = n & 63;
        const float bb = bias[n];
        for (int j = 0; j < 8; ++j) {
            const int m = m0 + j + lh * 8;
            const int b = m >> 10, l = m & 1023;
            const float v = c[t][j] + bb;
            size_t idx;
            if (mode < 2) idx = ((size_t)(b * 16 + hh) * 1024 + l) * 64 + d; // [b,h,l,d]
            else          idx = ((size_t)(b * 16 + hh) * 64 + d) * 1024 + l; // [b,h,d,l]
            out[idx] = (_Float16)v;
        }
    }
}

// ---------------------------------------------------------------------------
// Row norms ||q||^2, ||k||^2 over Dh=64 (f16 in, f32 out).
// ---------------------------------------------------------------------------
__global__ void hk_norm_kernel(const _Float16* __restrict__ Q,
                               const _Float16* __restrict__ K,
                               float* __restrict__ qn,
                               float* __restrict__ kn) {
    const int idx = blockIdx.x * blockDim.x + threadIdx.x; // 0 .. 131071
    const bool isK = idx >= 65536;
    const int r = idx & 65535;
    const _Float16* src = (isK ? K : Q) + (size_t)r * 64;
    float s = 0.f;
    for (int d = 0; d < 64; ++d) { float v = (float)src[d]; s += v * v; }
    (isK ? kn : qn)[r] = s;
}

// ---------------------------------------------------------------------------
// Flash-style hyperbolic attention. Grid = (B*H, L/64). Block = 128 (4 waves);
// each wave owns 16 query rows. K/V tiles (shared by all 4 waves) are staged
// global->LDS with async b128 copies, double-buffered: tile i+1 streams in
// while WMMAs + score math consume tile i.
// Score: exp(-log(u)) == 1/u with u = arg + sqrt(max(arg^2-1, 1e-8)).
// ---------------------------------------------------------------------------
__global__ void hk_attn_kernel(const _Float16* __restrict__ Q,
                               const _Float16* __restrict__ K,
                               const _Float16* __restrict__ Vt,
                               const float* __restrict__ qn,
                               const float* __restrict__ kn,
                               const float* __restrict__ coher,
                               _Float16* __restrict__ ctx) {
    __shared__ _Float16 kbuf[2][32 * KS];   // [kpos][d], padded rows
    __shared__ _Float16 vbuf[2][64 * VS];   // [d][kpos], padded rows
    __shared__ _Float16 pbuf[4][16 * PS];   // per-wave P staging

    const int bh = blockIdx.x;               // 0..63
    const int b = bh >> 4, h = bh & 15;
    const int tid = threadIdx.x;
    const int wave = tid >> 5;
    const int lane = tid & 31;
    const int lm = lane & 15, lh = lane >> 4;
    const int q0 = blockIdx.y * 64 + wave * 16;

    const _Float16* Qb = Q  + (size_t)bh * 1024 * 64;
    const _Float16* Kb = K  + (size_t)bh * 1024 * 64;
    const _Float16* Vb = Vt + (size_t)bh * 64 * 1024;
    const float* qnb = qn + bh * 1024;
    const float* knb = kn + bh * 1024;
    const float coh = coher[bh];

    // Cooperative tile stage: K tile 32x64 halves (256 x 16B chunks),
    // V tile 64x32 halves (256 x 16B chunks); 128 threads, 2 chunks each.
    auto stage = [&](int buf, int k0) {
        for (int q = tid; q < 256; q += 128) {
            const int r = q >> 3, c = q & 7;
            hk_cp16(Kb + (size_t)(k0 + r) * 64 + c * 8, &kbuf[buf][r * KS + c * 8]);
        }
        for (int q = tid; q < 256; q += 128) {
            const int d = q >> 2, c = q & 3;
            hk_cp16(Vb + (size_t)d * 1024 + k0 + c * 8, &vbuf[buf][d * VS + c * 8]);
        }
    };

    // Q as two A-layout chunks (Dh = 64 = 2 * K32).
    v16h qa[2];
    {
        const _Float16* p = Qb + (size_t)(q0 + lm) * 64;
        for (int c = 0; c < 2; ++c) {
            const int dbase = 32 * c + lh * 16;
            for (int i = 0; i < 8; ++i) {
                qa[c][i]     = p[dbase + i];
                qa[c][8 + i] = p[dbase + 8 + i];
            }
        }
    }

    // Per-lane row norms (rows of this lane's C elements: m = j + lh*8).
    float qn_r[8], om[8];
    for (int j = 0; j < 8; ++j) {
        const float v = qnb[q0 + j + lh * 8];
        qn_r[j] = v;
        om[j] = 1.0f - fminf(v, 0.99f);
    }

    float m_i[8], l_i[8];
    v8f acc[4];
    for (int j = 0; j < 8; ++j) { m_i[j] = -1e30f; l_i[j] = 0.f; }
    for (int t = 0; t < 4; ++t)
        for (int j = 0; j < 8; ++j) acc[t][j] = 0.f;

    _Float16* plds = pbuf[wave];

    stage(0, 0);   // prefetch tile 0

    for (int it = 0; it < 32; ++it) {
        const int k0 = it * 32;
        hk_async_wait();
        __syncthreads();                      // tile `it` visible to all waves
        if (it + 1 < 32) stage((it + 1) & 1, k0 + 32);
        const _Float16* kb_l = kbuf[it & 1];
        const _Float16* vb_l = vbuf[it & 1];

        // ---- S = Q * K^T (16 rows x 32 key columns) ----
        v8f s_acc[2];
        for (int t = 0; t < 2; ++t)
            for (int j = 0; j < 8; ++j) s_acc[t][j] = 0.f;
        for (int t = 0; t < 2; ++t) {
            const int rr = t * 16 + lm;       // kpos row within tile
            for (int c = 0; c < 2; ++c) {
                v16h kmat;
                const _Float16* p = kb_l + rr * KS + 32 * c + lh * 16;
                for (int i = 0; i < 16; ++i) kmat[i] = p[i];
                s_acc[t] = __builtin_amdgcn_wmma_f32_16x16x32_f16(
                    false, qa[c], false, kmat, (short)0, s_acc[t], false, false);
            }
        }

        // ---- hyperbolic distance -> score, online softmax ----
        float kn_t[2], on_t[2];
        for (int t = 0; t < 2; ++t) {
            const float v = knb[k0 + t * 16 + lm];
            kn_t[t] = v;
            on_t[t] = 1.0f - fminf(v, 0.99f);
        }
        float pvals[2][8];
        float mrow[8];
        for (int j = 0; j < 8; ++j) {
            float mx = -1e30f;
            for (int t = 0; t < 2; ++t) {
                const float dot   = s_acc[t][j];
                const float diff  = qn_r[j] + kn_t[t] - 2.0f * dot;
                const float denom = om[j] * on_t[t];
                const float arg   = 1.0f + 2.0f * diff / (denom + 1e-8f);
                const float u     = arg + sqrtf(fmaxf(arg * arg - 1.0f, 1e-8f));
                const float sc    = coh / u;  // == exp(-acosh-dist) * coherence
                pvals[t][j] = sc;
                mx = fmaxf(mx, sc);
            }
            for (int o = 1; o < 16; o <<= 1)
                mx = fmaxf(mx, __shfl_xor(mx, o, 32));
            mrow[j] = mx;
        }
        for (int j = 0; j < 8; ++j) {
            const float mnew  = fmaxf(m_i[j], mrow[j]);
            const float scale = expf(m_i[j] - mnew);
            float rs = 0.f;
            for (int t = 0; t < 2; ++t) {
                const float p = expf(pvals[t][j] - mnew);
                pvals[t][j] = p;
                rs += p;
            }
            for (int o = 1; o < 16; o <<= 1)
                rs += __shfl_xor(rs, o, 32);
            l_i[j] = l_i[j] * scale + rs;
            m_i[j] = mnew;
            for (int t = 0; t < 4; ++t) acc[t][j] *= scale;
        }

        // ---- P relayout through LDS: C-layout -> row-major [16][PS] ----
        for (int t = 0; t < 2; ++t)
            for (int j = 0; j < 8; ++j) {
                const int r = j + lh * 8;
                plds[r * PS + t * 16 + lm] = (_Float16)pvals[t][j];
            }
        __syncthreads();

        v16h pa; // A-layout read of P (K-dim = 32 key positions)
        {
            const _Float16* p = plds + lm * PS + lh * 16;
            for (int i = 0; i < 8; ++i) {
                pa[i]     = p[i];
                pa[8 + i] = p[8 + i];
            }
        }
        __syncthreads();

        // ---- acc += P * V (4 N-subtiles of Dh) ----
        for (int ds = 0; ds < 4; ++ds) {
            v16h vmat;
            const _Float16* p = vb_l + (ds * 16 + lm) * VS + lh * 16;
            for (int i = 0; i < 16; ++i) vmat[i] = p[i];
            acc[ds] = __builtin_amdgcn_wmma_f32_16x16x32_f16(
                false, pa, false, vmat, (short)0, acc[ds], false, false);
        }
    }

    // ---- normalize and write context in merged-head layout [b, l, h*64+d] ----
    for (int ds = 0; ds < 4; ++ds) {
        const int d = h * 64 + ds * 16 + lm;
        for (int j = 0; j < 8; ++j) {
            const int r = q0 + j + lh * 8;
            const float v = acc[ds][j] / l_i[j];
            ctx[((size_t)b * 1024 + r) * 1024 + d] = (_Float16)v;
        }
    }
}

// ---------------------------------------------------------------------------
// Output projection: out[m,n] = sum_k ctx[m,k] * Wo[n,k] + bo[n]  (f32 out)
// ---------------------------------------------------------------------------
__global__ void hk_gemm_out(const _Float16* __restrict__ Ctx,
                            const float* __restrict__ Wo,
                            const float* __restrict__ bo,
                            float* __restrict__ Out) {
    const int wave = threadIdx.x >> 5;
    const int lane = threadIdx.x & 31;
    const int lm = lane & 15, lh = lane >> 4;
    const int m0 = blockIdx.y * 64 + wave * 16;
    const int n0 = blockIdx.x * 64;

    v8f c[4];
    for (int t = 0; t < 4; ++t)
        for (int j = 0; j < 8; ++j) c[t][j] = 0.0f;

    for (int k0 = 0; k0 < 1024; k0 += 32) {
        v16h a;
        const _Float16* px = Ctx + (size_t)(m0 + lm) * 1024 + k0 + lh * 16;
        for (int i = 0; i < 8; ++i) {
            a[i]     = px[i];
            a[8 + i] = px[8 + i];
        }
        for (int t = 0; t < 4; ++t) {
            v16h bm;
            const float* pw = Wo + (size_t)(n0 + t * 16 + lm) * 1024 + k0 + lh * 16;
            for (int i = 0; i < 16; ++i) bm[i] = (_Float16)pw[i];
            c[t] = __builtin_amdgcn_wmma_f32_16x16x32_f16(
                false, a, false, bm, (short)0, c[t], false, false);
        }
    }

    for (int t = 0; t < 4; ++t) {
        const int n = n0 + t * 16 + lm;
        const float bb = bo[n];
        for (int j = 0; j < 8; ++j) {
            const int m = m0 + j + lh * 8;
            Out[(size_t)m * 1024 + n] = c[t][j] + bb;
        }
    }
}

// ---------------------------------------------------------------------------
extern "C" void kernel_launch(void* const* d_in, const int* in_sizes, int n_in,
                              void* d_out, int out_size, void* d_ws, size_t ws_size,
                              hipStream_t stream) {
    const float* hs   = (const float*)d_in[0];
    const float* Wq   = (const float*)d_in[1];
    const float* bq   = (const float*)d_in[2];
    const float* Wk   = (const float*)d_in[3];
    const float* bk   = (const float*)d_in[4];
    const float* Wv   = (const float*)d_in[5];
    const float* bv   = (const float*)d_in[6];
    const float* Wo   = (const float*)d_in[7];
    const float* bo   = (const float*)d_in[8];
    const float* base = (const float*)d_in[9];
    const float* natf = (const float*)d_in[10];
    const float* ph0  = (const float*)d_in[11];

    char* ws = (char*)d_ws;
    _Float16* Qf  = (_Float16*)(ws);
    _Float16* Kf  = (_Float16*)(ws + (size_t)8  * 1024 * 1024);
    _Float16* Vtf = (_Float16*)(ws + (size_t)16 * 1024 * 1024);
    _Float16* Ctx = (_Float16*)(ws + (size_t)24 * 1024 * 1024);
    float* qn     = (float*)   (ws + (size_t)32 * 1024 * 1024);
    float* kn     = (float*)   (ws + (size_t)32 * 1024 * 1024 + 256 * 1024);
    float* coher  = (float*)   (ws + (size_t)32 * 1024 * 1024 + 512 * 1024);

    float* out        = (float*)d_out;
    float* out_phases = out + (size_t)4 * 1024 * 1024; // [4,16]
    float* out_order  = out_phases + 64;               // [4]

    hk_phase_kernel<<<dim3(1), dim3(64), 0, stream>>>(
        base, natf, ph0, out_phases, out_order, coher);

    hk_gemm_qkv<<<dim3(16, 64, 3), dim3(128), 0, stream>>>(
        hs, Wq, bq, Wk, bk, Wv, bv, Qf, Kf, Vtf);

    hk_norm_kernel<<<dim3(512), dim3(256), 0, stream>>>(Qf, Kf, qn, kn);

    hk_attn_kernel<<<dim3(64, 16), dim3(128), 0, stream>>>(
        Qf, Kf, Vtf, qn, kn, coher, Ctx);

    hk_gemm_out<<<dim3(16, 64), dim3(128), 0, stream>>>(Ctx, Wo, bo, out);
}
